// SelfAttention_43997644980711
// MI455X (gfx1250) — matrix operand order repaired and checked by reading, not compiled, not run on hardware
//
#include <hip/hip_runtime.h>

#define LSEQ  512
#define NHEAD 16
#define HD    64
#define ODIM  1024
#define XB    4
#define QKSCALE 0.125f
#define VTSTRIDE 40   // 32 + 8 pad, keeps 16B alignment (80B row stride)

// Tensor Data Mover path (guarded; falls back to synchronous staging)
#if defined(__HIP_DEVICE_COMPILE__) && __has_builtin(__builtin_amdgcn_tensor_load_to_lds)
#define USE_TDM 1
#if __has_include(<hip/amd_detail/amd_gfx1250_TDM.h>)
#define TDM_ARGS6 1
#endif
#endif

typedef __attribute__((ext_vector_type(16))) __bf16 v16bf;
typedef __attribute__((ext_vector_type(2)))  __bf16 v2bf;
typedef __attribute__((ext_vector_type(8)))  float  v8f;
typedef __attribute__((ext_vector_type(4)))  unsigned int u32x4;
typedef __attribute__((ext_vector_type(8)))  int i32x8;
typedef __attribute__((ext_vector_type(4)))  int i32x4;

union Frag {
  uint4          q[2];
  unsigned int   w[8];
  unsigned short u[16];
  v16bf          v;
};

__device__ __forceinline__ unsigned int pack2(float a, float b) {
#if __has_builtin(__builtin_amdgcn_cvt_pk_bf16_f32)
  v2bf p = __builtin_amdgcn_cvt_pk_bf16_f32(a, b);   // v_cvt_pk_bf16_f32
  return __builtin_bit_cast(unsigned int, p);
#else
  v2bf p = {(__bf16)a, (__bf16)b};
  return __builtin_bit_cast(unsigned int, p);
#endif
}

__device__ __forceinline__ unsigned short f2bf(float f) {
  __bf16 h = (__bf16)f;
  return __builtin_bit_cast(unsigned short, h);
}

// ---------------------------------------------------------------------------
// Phase 0: one-shot f32 -> bf16 conversion of the three 64x1024 weight mats.
// ---------------------------------------------------------------------------
__global__ __launch_bounds__(256) void cvt_w_kernel(const float* __restrict__ w0,
                                                    const float* __restrict__ w1,
                                                    const float* __restrict__ w2,
                                                    unsigned short* __restrict__ o0,
                                                    unsigned short* __restrict__ o1,
                                                    unsigned short* __restrict__ o2) {
  const float* w = (blockIdx.y == 0) ? w0 : ((blockIdx.y == 1) ? w1 : w2);
  unsigned short* o = (blockIdx.y == 0) ? o0 : ((blockIdx.y == 1) ? o1 : o2);
  const int i = blockIdx.x * blockDim.x + threadIdx.x;   // one float4 each
  float4 f = ((const float4*)w)[i];
  uint2 r;
  r.x = pack2(f.x, f.y);
  r.y = pack2(f.z, f.w);
  ((uint2*)o)[i] = r;
}

// ---------------------------------------------------------------------------
// Phase 1: fused projection  y = x @ W^T + b, written as bf16 into the folded
// [X, NHEAD, LSEQ, HD] layout. Row mapping (reshape/transpose view trick):
//   r = b*512 + l  ->  x = b/16, l2 = (b%16)*32 + l/16, m = l%16
// A 16-row tile (r0 multiple of 16) keeps (x, l2) fixed with m = i + 8*half.
// ---------------------------------------------------------------------------
struct ProjArgs {
  const float* in;          // [32768, 1024] f32
  const unsigned short* w;  // [64, 1024] bf16 (pre-converted)
  const float* b;           // [64]
  unsigned short* out;      // bf16 [X, NHEAD, LSEQ, HD]
};

__global__ __launch_bounds__(256) void proj_kernel(ProjArgs A0, ProjArgs A1, ProjArgs A2) {
  ProjArgs P = (blockIdx.y == 0) ? A0 : ((blockIdx.y == 1) ? A1 : A2);

  const int t      = threadIdx.x & 31;
  const int wv     = threadIdx.x >> 5;
  const int half   = t >> 4;       // lane group selects K-half per ISA layouts
  const int lane16 = t & 15;
  const int r0     = blockIdx.x * 128 + wv * 16;

  const float* inrow = P.in + (size_t)(r0 + lane16) * ODIM;

  v8f z = {0.f, 0.f, 0.f, 0.f, 0.f, 0.f, 0.f, 0.f};
  v8f acc[4];
  #pragma unroll
  for (int n = 0; n < 4; ++n) acc[n] = z;

  for (int kb = 0; kb < ODIM; kb += 32) {
    __builtin_prefetch(inrow + kb + 256, 0, 1);  // global_prefetch_b8, stream ahead

    // A fragment 16x32 bf16: lane<16 holds K {0..7,16..23}, lane>=16 {8..15,24..31}
    Frag a;
    {
      const float4* p0 = (const float4*)(inrow + kb + 8 * half);
      float4 f0 = p0[0], f1 = p0[1];
      const float4* p1 = (const float4*)(inrow + kb + 16 + 8 * half);
      float4 f2 = p1[0], f3 = p1[1];
      a.w[0] = pack2(f0.x, f0.y); a.w[1] = pack2(f0.z, f0.w);
      a.w[2] = pack2(f1.x, f1.y); a.w[3] = pack2(f1.z, f1.w);
      a.w[4] = pack2(f2.x, f2.y); a.w[5] = pack2(f2.z, f2.w);
      a.w[6] = pack2(f3.x, f3.y); a.w[7] = pack2(f3.z, f3.w);
    }

    // B fragments 32x16 bf16 straight from pre-converted weights: 2 b128 loads
    #pragma unroll
    for (int n = 0; n < 4; ++n) {
      Frag bf;
      const unsigned short* wrow =
          P.w + (size_t)(n * 16 + lane16) * ODIM + kb + 8 * half;
      bf.q[0] = *(const uint4*)wrow;
      bf.q[1] = *(const uint4*)(wrow + 16);
      acc[n] = __builtin_amdgcn_wmma_f32_16x16x32_bf16(
          false, a.v, false, bf.v, (short)0, acc[n], false, false);
    }
  }

  // bias + store into folded bf16 layout (index math hoisted out of the loops)
  const int bo = r0 >> 9;
  const int l0 = r0 & 511;                      // multiple of 16
  const int x  = bo >> 4;
  const int l2 = ((bo & 15) << 5) | (l0 >> 4);
  unsigned short* outb =
      P.out + ((size_t)x * NHEAD * LSEQ + l2) * HD;  // head stride = LSEQ*HD
  #pragma unroll
  for (int n = 0; n < 4; ++n) {
    const int col = n * 16 + lane16;
    const float bias = P.b[col];
    #pragma unroll
    for (int i = 0; i < 8; ++i) {
      const size_t idx = (size_t)(i + 8 * half) * (LSEQ * HD) + col;
      outb[idx] = f2bf(acc[n][i] + bias);
    }
  }
}

// ---------------------------------------------------------------------------
// Phase 2: flash-style attention per (x, head). 8 waves/block, 16 queries/wave.
// K tile staged via the Tensor Data Mover when available (one descriptor per
// 32x64 bf16 tile, overlapped with V staging, completion via s_wait_tensorcnt);
// V is staged transposed so A*V B-fragments are contiguous ds_load_b128s.
// ---------------------------------------------------------------------------
__global__ __launch_bounds__(256) void attn_kernel(
    const unsigned short* __restrict__ Qp,
    const unsigned short* __restrict__ Kp,
    const unsigned short* __restrict__ Vp,
    const int* __restrict__ Qlen,
    const int* __restrict__ Vlen,
    float* __restrict__ out) {
  __shared__ unsigned short KS[32 * 64];          // K block, row-major
  __shared__ unsigned short VT[64 * VTSTRIDE];    // V block, transposed [d][k]
  __shared__ unsigned short PT[8 * 512];          // per-wave P^T scratch [k][q]

  const int qblk = blockIdx.x;     // 0..3
  const int m    = blockIdx.y;     // head
  const int x    = blockIdx.z;     // folded batch
  const int tid  = threadIdx.x;
  const int t      = tid & 31;
  const int wv     = tid >> 5;
  const int half   = t >> 4;
  const int lane16 = t & 15;
  const int q0     = qblk * 128 + wv * 16;
  const int q0wg   = qblk * 128;
  const int kv     = Vlen[x];
  const int ql     = Qlen[x];

  const size_t headoff = ((size_t)x * NHEAD + m) * (size_t)LSEQ * HD;
  const unsigned short* Qh = Qp + headoff;
  const unsigned short* Kh = Kp + headoff;
  const unsigned short* Vh = Vp + headoff;

  // Q A-fragments (16q x 32d), two d-halves; contiguous 16B runs per lane
  Frag aq[2];
  #pragma unroll
  for (int f = 0; f < 2; ++f) {
    const unsigned short* qr = Qh + (size_t)(q0 + lane16) * HD + 32 * f + 8 * half;
    aq[f].q[0] = *(const uint4*)qr;
    aq[f].q[1] = *(const uint4*)(qr + 16);
  }

  v8f z = {0.f, 0.f, 0.f, 0.f, 0.f, 0.f, 0.f, 0.f};
  v8f o[4];
  #pragma unroll
  for (int n = 0; n < 4; ++n) o[n] = z;
  float mrow[8], lrow[8];
  #pragma unroll
  for (int i = 0; i < 8; ++i) { mrow[i] = -3.0e38f; lrow[i] = 0.f; }

  const int kend = (kv < q0wg + 128) ? kv : (q0wg + 128);  // keys past this fully masked
  const int nkb  = (kend + 31) >> 5;                       // uniform across the block

  for (int kb = 0; kb < nkb; ++kb) {
    const int k_base = kb << 5;
    __syncthreads();

#if defined(USE_TDM)
    // --- K tile via Tensor Data Mover: one 2D descriptor, issued by wave 0 ---
    if (wv == 0) {
      unsigned long long ga =
          (unsigned long long)(Kh + (size_t)k_base * HD);       // byte address
      unsigned int lds = (unsigned int)(unsigned long long)(&KS[0]);
      u32x4 g0 = {
          1u,                                                   // count=1 (valid D#)
          lds,                                                  // lds_addr [63:32]
          (unsigned int)(ga & 0xffffffffu),                     // global_addr lo
          (unsigned int)((ga >> 32) & 0x01ffffffu) | (2u << 30) // addr hi | type=2
      };
      i32x8 g1 = {
          (int)(1u << 16),          // data_size=1 -> 2 bytes
          (int)(64u << 16),         // tensor_dim0 = 64  (bits 79:48)
          (int)(512u << 16),        // tensor_dim1 = 512 (bits 111:80)
          (int)(64u << 16),         // tile_dim0 = 64    (bits 127:112)
          (int)32,                  // tile_dim1 = 32    (bits 143:128)
          (int)64,                  // tensor_dim0_stride = 64 (bits 207:160)
          0, 0
      };
      i32x4 zg = {0, 0, 0, 0};
#if defined(TDM_ARGS6)
      i32x8 z8 = {0, 0, 0, 0, 0, 0, 0, 0};
      __builtin_amdgcn_tensor_load_to_lds(g0, g1, zg, zg, z8, 0);
#else
      __builtin_amdgcn_tensor_load_to_lds(g0, g1, zg, zg, 0);
#endif
    }
#endif

    {  // V tile: cooperative load + transpose into VT (overlaps the TDM copy)
      const int rowk = tid >> 3;          // 0..31
      const int cseg = (tid & 7) << 3;    // 0..56
#if !defined(USE_TDM)
      uint4 kq = *(const uint4*)(Kh + (size_t)(k_base + rowk) * HD + cseg);
      *(uint4*)&KS[rowk * 64 + cseg] = kq;
#endif
      uint4 vq = *(const uint4*)(Vh + (size_t)(k_base + rowk) * HD + cseg);
      unsigned short vs[8];
      vs[0] = vq.x & 0xffff; vs[1] = vq.x >> 16;
      vs[2] = vq.y & 0xffff; vs[3] = vq.y >> 16;
      vs[4] = vq.z & 0xffff; vs[5] = vq.z >> 16;
      vs[6] = vq.w & 0xffff; vs[7] = vq.w >> 16;
      #pragma unroll
      for (int j = 0; j < 8; ++j) VT[(cseg + j) * VTSTRIDE + rowk] = vs[j];
    }

#if defined(USE_TDM)
    if (wv == 0) __builtin_amdgcn_s_wait_tensorcnt(0);  // TDM done before barrier
#endif
    __syncthreads();

    if (k_base > q0 + 15) continue;       // wave-uniform causal skip

    // S = Q K^T for this 16q x 32k block: two 16x16 accumulators, 4 WMMAs
    v8f sacc[2];
    #pragma unroll
    for (int ks = 0; ks < 2; ++ks) {
      v8f s = z;
      #pragma unroll
      for (int fd = 0; fd < 2; ++fd) {
        Frag bk;
        const unsigned short* kp = &KS[(16 * ks + lane16) * 64 + 32 * fd + 8 * half];
        bk.q[0] = *(const uint4*)kp;
        bk.q[1] = *(const uint4*)(kp + 16);
        s = __builtin_amdgcn_wmma_f32_16x16x32_bf16(
            false, aq[fd].v, false, bk.v, (short)0, s, false, false);
      }
      sacc[ks] = s;
    }

    // scale + causal + key-length masks, online softmax update
    float p0[8], p1[8], al[8], r[8];
    const int kk0 = k_base + lane16;
    const int kk1 = kk0 + 16;
    #pragma unroll
    for (int i = 0; i < 8; ++i) {
      const int q = q0 + i + 8 * half;
      float v0 = ((kk0 > q) || (kk0 >= kv)) ? -1e30f : sacc[0][i] * QKSCALE;
      float v1 = ((kk1 > q) || (kk1 >= kv)) ? -1e30f : sacc[1][i] * QKSCALE;
      p0[i] = v0; p1[i] = v1;
      r[i]  = fmaxf(v0, v1);
    }
    #pragma unroll
    for (int msk = 1; msk <= 8; msk <<= 1)
      #pragma unroll
      for (int i = 0; i < 8; ++i)
        r[i] = fmaxf(r[i], __shfl_xor(r[i], msk, 32));
    #pragma unroll
    for (int i = 0; i < 8; ++i) {
      const float mn = fmaxf(mrow[i], r[i]);
      al[i]   = __expf(mrow[i] - mn);
      mrow[i] = mn;
      p0[i]   = __expf(p0[i] - mn);
      p1[i]   = __expf(p1[i] - mn);
    }
    float rs[8];
    #pragma unroll
    for (int i = 0; i < 8; ++i) rs[i] = p0[i] + p1[i];
    #pragma unroll
    for (int msk = 1; msk <= 8; msk <<= 1)
      #pragma unroll
      for (int i = 0; i < 8; ++i)
        rs[i] += __shfl_xor(rs[i], msk, 32);
    #pragma unroll
    for (int i = 0; i < 8; ++i) lrow[i] = lrow[i] * al[i] + rs[i];
    #pragma unroll
    for (int n = 0; n < 4; ++n)
      #pragma unroll
      for (int i = 0; i < 8; ++i) o[n][i] *= al[i];

    // transpose P through per-wave LDS into A-fragment layout
    unsigned short* ptw = &PT[wv * 512];
    {
      uint4 pk;
      pk.x = pack2(p0[0], p0[1]); pk.y = pack2(p0[2], p0[3]);
      pk.z = pack2(p0[4], p0[5]); pk.w = pack2(p0[6], p0[7]);
      *(uint4*)&ptw[lane16 * 16 + 8 * half] = pk;
      pk.x = pack2(p1[0], p1[1]); pk.y = pack2(p1[2], p1[3]);
      pk.z = pack2(p1[4], p1[5]); pk.w = pack2(p1[6], p1[7]);
      *(uint4*)&ptw[(16 + lane16) * 16 + 8 * half] = pk;
    }
    asm volatile("s_wait_dscnt 0" ::: "memory");
    Frag pa;
    #pragma unroll
    for (int e = 0; e < 8; ++e) {
      pa.u[e]     = ptw[(8 * half + e) * 16 + lane16];
      pa.u[e + 8] = ptw[(16 + 8 * half + e) * 16 + lane16];
    }

    // O += P V : 4 WMMAs, V B-fragments contiguous from transposed LDS tile
    #pragma unroll
    for (int n = 0; n < 4; ++n) {
      Frag bv;
      const unsigned short* vp = &VT[(16 * n + lane16) * VTSTRIDE + 8 * half];
      bv.q[0] = *(const uint4*)vp;
      bv.q[1] = *(const uint4*)(vp + 16);
      o[n] = __builtin_amdgcn_wmma_f32_16x16x32_bf16(
          false, pa.v, false, bv.v, (short)0, o[n], false, false);
    }
  }

  // finalize: 1/l, Q_len mask, write [X, L, M*D] f32
  float fac[8];
  #pragma unroll
  for (int i = 0; i < 8; ++i) {
    const int q = q0 + i + 8 * half;
    fac[i] = (q < ql) ? (1.0f / lrow[i]) : 0.0f;
  }
  float* ob = out + (size_t)x * LSEQ * ODIM + (size_t)m * HD;
  #pragma unroll
  for (int n = 0; n < 4; ++n) {
    const int col = n * 16 + lane16;
    #pragma unroll
    for (int i = 0; i < 8; ++i) {
      const int q = q0 + i + 8 * half;
      ob[(size_t)q * ODIM + col] = o[n][i] * fac[i];
    }
  }
}

// ---------------------------------------------------------------------------
extern "C" void kernel_launch(void* const* d_in, const int* in_sizes, int n_in,
                              void* d_out, int out_size, void* d_ws, size_t ws_size,
                              hipStream_t stream) {
  const float* Qs  = (const float*)d_in[0];
  const float* Ks  = (const float*)d_in[1];
  const float* Vs  = (const float*)d_in[2];
  const int*   Qlen = (const int*)d_in[3];
  const int*   Vlen = (const int*)d_in[4];
  const float* WQw = (const float*)d_in[5];
  const float* WQb = (const float*)d_in[6];
  const float* WKw = (const float*)d_in[7];
  const float* WKb = (const float*)d_in[8];
  const float* WVw = (const float*)d_in[9];
  const float* WVb = (const float*)d_in[10];

  const size_t per  = (size_t)XB * NHEAD * LSEQ * HD;  // 2,097,152 bf16 elems
  const size_t wsz  = (size_t)HD * ODIM;               // 65,536 elems per weight
  unsigned short* ws = (unsigned short*)d_ws;
  unsigned short* Qp = ws;
  unsigned short* Kp = ws + per;
  unsigned short* Vp = ws + 2 * per;
  unsigned short* Wq = ws + 3 * per;
  unsigned short* Wk = Wq + wsz;
  unsigned short* Wv = Wk + wsz;

  // weights: 65536 f32 each, one float4 per thread -> 64 blocks of 256
  cvt_w_kernel<<<dim3(64, 3, 1), 256, 0, stream>>>(WQw, WKw, WVw, Wq, Wk, Wv);

  ProjArgs aq{Qs, Wq, WQb, Qp};
  ProjArgs ak{Ks, Wk, WKb, Kp};
  ProjArgs av{Vs, Wv, WVb, Vp};
  proj_kernel<<<dim3(256, 3, 1), 256, 0, stream>>>(aq, ak, av);

  attn_kernel<<<dim3(LSEQ / 128, NHEAD, XB), 256, 0, stream>>>(
      Qp, Kp, Vp, Qlen, Vlen, (float*)d_out);
}
